// SequenceBlock_75256416960729
// MI455X (gfx1250) — matrix operand order repaired and checked by reading, not compiled, not run on hardware
//
#include <hip/hip_runtime.h>
#include <cstdint>
#include <cstddef>

// Problem constants (match reference)
#define BB 16
#define LL 2048
#define HH 512
#define NN 64
#define MM (BB * LL)   // 32768 rows for the output GEMM

typedef __attribute__((ext_vector_type(8)))  float  v8f;
typedef __attribute__((ext_vector_type(16))) __bf16 v16bf;
typedef __attribute__((ext_vector_type(8)))  __bf16 v8bf;
typedef __attribute__((ext_vector_type(4)))  unsigned tdm_g0_t;  // uint32x4
typedef __attribute__((ext_vector_type(8)))  int      tdm_g1_t;  // int32x8
typedef __attribute__((ext_vector_type(4)))  int      tdm_g2_t;  // int32x4

static __device__ __forceinline__ unsigned short f32_to_bf16_rn(float f) {
  unsigned u = __builtin_bit_cast(unsigned, f);
  unsigned r = u + 0x7FFFu + ((u >> 16) & 1u);   // round-to-nearest-even
  return (unsigned short)(r >> 16);
}
static __device__ __forceinline__ float bf16_bits_to_f32(unsigned short h) {
  return __builtin_bit_cast(float, (unsigned)h << 16);
}

// Issue one TDM 2D tile load: global (bf16 elements, row-major, row stride
// `tensorDim0` elements) -> LDS, tile = tileRows x tileCols elements.
static __device__ __forceinline__ void tdm_load_2d_bf16(
    const unsigned short* gptr, unsigned ldsByteOff,
    unsigned tensorDim0, unsigned tensorDim1,
    unsigned tileCols, unsigned tileRows)
{
  unsigned long long ga = (unsigned long long)(uintptr_t)gptr;
  tdm_g0_t g0;
  g0[0] = 1u;                                        // count=1, user mode
  g0[1] = ldsByteOff;                                // lds_addr
  g0[2] = (unsigned)(ga & 0xFFFFFFFFu);              // global_addr[31:0]
  g0[3] = (unsigned)((ga >> 32) & 0x1FFFFFFu) | (2u << 30);  // addr[56:32] | type=2

  tdm_g1_t g1;
  g1[0] = (int)(1u << 16);                           // workgroup_mask=0, data_size=1 (2B)
  g1[1] = (int)((tensorDim0 & 0xFFFFu) << 16);       // tensor_dim0[15:0]
  g1[2] = (int)((tensorDim0 >> 16) | ((tensorDim1 & 0xFFFFu) << 16));
  g1[3] = (int)((tensorDim1 >> 16) | (tileCols << 16));   // tile_dim0
  g1[4] = (int)(tileRows & 0xFFFFu);                 // tile_dim1 (tile_dim2=0)
  g1[5] = (int)tensorDim0;                           // tensor_dim0_stride[31:0]
  g1[6] = 0;                                         // stride0 hi | stride1 lo
  g1[7] = 0;                                         // stride1 hi

  tdm_g2_t gz = {0, 0, 0, 0};                        // 2D tensor: groups 2/3 unused
#if __clang_major__ >= 23
  tdm_g1_t gz8 = {0, 0, 0, 0, 0, 0, 0, 0};
  __builtin_amdgcn_tensor_load_to_lds(g0, g1, gz, gz, gz8, 0);
#else
  __builtin_amdgcn_tensor_load_to_lds(g0, g1, gz, gz, 0);
#endif
}

// ---------------------------------------------------------------------------
// Kernel 1: per-(h,n) discretized SSM parameters.
//   a = exp(dt*Lambda)  (complex),  w = 2 * C * (exp(dt*Lambda)-1)/Lambda
// ---------------------------------------------------------------------------
__global__ __launch_bounds__(256) void s4d_params_kernel(
    const float* __restrict__ Lre, const float* __restrict__ Lim,
    const float* __restrict__ log_dt,
    const float* __restrict__ Cre, const float* __restrict__ Cim,
    float* __restrict__ aRe, float* __restrict__ aIm,
    float* __restrict__ wRe, float* __restrict__ wIm)
{
  int t = blockIdx.x * 256 + threadIdx.x;   // < H*N = 32768
  int h = t >> 6;
  int n = t & (NN - 1);
  float dt = expf(log_dt[h]);
  float lr = Lre[n], li = Lim[n];
  float dr = dt * lr, di = dt * li;
  float er = expf(dr);
  float sn, cs;
  sincosf(di, &sn, &cs);
  float ar = er * cs, ai = er * sn;          // a = exp(dtA)
  float em1r = ar - 1.0f, em1i = ai;         // exp(dtA) - 1
  float inv = 1.0f / (lr * lr + li * li);
  float dBr = (em1r * lr + em1i * li) * inv; // (e-1)/Lambda
  float dBi = (em1i * lr - em1r * li) * inv;
  float cr = Cre[t], ci = Cim[t];
  aRe[t] = ar;
  aIm[t] = ai;
  wRe[t] = 2.0f * (cr * dBr - ci * dBi);
  wIm[t] = 2.0f * (cr * dBi + ci * dBr);
}

// ---------------------------------------------------------------------------
// Kernel 2: transpose W_out [K=H, N=H] -> [N, K] and split into bf16 hi/lo.
// ---------------------------------------------------------------------------
__global__ __launch_bounds__(256) void wsplit_kernel(
    const float* __restrict__ W,
    unsigned short* __restrict__ Wh, unsigned short* __restrict__ Wl)
{
  int t = blockIdx.x * 256 + threadIdx.x;   // < H*H
  int n = t >> 9;
  int k = t & (HH - 1);
  float f = W[(size_t)k * HH + n];
  unsigned short hb = f32_to_bf16_rn(f);
  float rem = f - bf16_bits_to_f32(hb);
  size_t o = (size_t)n * HH + k;
  Wh[o] = hb;
  Wl[o] = f32_to_bf16_rn(rem);
}

// ---------------------------------------------------------------------------
// Kernel 3: LayerNorm over H. One wave32 per row (b,l); 8 rows per block.
// ---------------------------------------------------------------------------
__global__ __launch_bounds__(256) void layernorm_kernel(
    const float* __restrict__ x,
    const float* __restrict__ scale, const float* __restrict__ bias,
    float* __restrict__ un)
{
  int row  = blockIdx.x * 8 + (threadIdx.x >> 5);  // < B*L
  int lane = threadIdx.x & 31;
  const float4* xr = (const float4*)(x + (size_t)row * HH);
  const float4* sc = (const float4*)scale;
  const float4* bi = (const float4*)bias;

  float4 v[4];
  float s = 0.0f, ss = 0.0f;
#pragma unroll
  for (int j = 0; j < 4; ++j) {
    v[j] = xr[j * 32 + lane];
    s  += v[j].x + v[j].y + v[j].z + v[j].w;
    ss += v[j].x * v[j].x + v[j].y * v[j].y + v[j].z * v[j].z + v[j].w * v[j].w;
  }
#pragma unroll
  for (int off = 16; off; off >>= 1) {
    s  += __shfl_xor(s, off, 32);
    ss += __shfl_xor(ss, off, 32);
  }
  float mu  = s * (1.0f / HH);
  float var = ss * (1.0f / HH) - mu * mu;
  float rin = rsqrtf(var + 1e-6f);

  float4* ur = (float4*)(un + (size_t)row * HH);
#pragma unroll
  for (int j = 0; j < 4; ++j) {
    int p = j * 32 + lane;
    float4 sv = sc[p], bv = bi[p], o;
    o.x = (v[j].x - mu) * rin * sv.x + bv.x;
    o.y = (v[j].y - mu) * rin * sv.y + bv.y;
    o.z = (v[j].z - mu) * rin * sv.z + bv.z;
    o.w = (v[j].w - mu) * rin * sv.w + bv.w;
    ur[p] = o;
  }
}

// ---------------------------------------------------------------------------
// Kernel 4: S4D recurrence scan + D*u skip + GELU (tanh approx).
// One wave32 per (b,h); each lane carries 2 complex states (n, n+32).
// Writes the activation pre-split into bf16 hi/lo planes for the WMMA GEMM.
// ---------------------------------------------------------------------------
__global__ __launch_bounds__(256) void s4d_scan_kernel(
    const float* __restrict__ u,      // u_norm [B, L, H]
    const float* __restrict__ aRe, const float* __restrict__ aIm,
    const float* __restrict__ wRe, const float* __restrict__ wIm,
    const float* __restrict__ Dp,
    unsigned short* __restrict__ yh,  // [M, K] bf16 hi plane of gelu(y)
    unsigned short* __restrict__ yl)  // [M, K] bf16 lo (residual) plane
{
  int wid  = blockIdx.x * 8 + (threadIdx.x >> 5);  // < B*H = 8192
  int b    = wid >> 9;
  int h    = wid & (HH - 1);
  int lane = threadIdx.x & 31;
  int p0 = h * NN + lane;
  int p1 = p0 + 32;

  float a0r = aRe[p0], a0i = aIm[p0], w0r = wRe[p0], w0i = wIm[p0];
  float a1r = aRe[p1], a1i = aIm[p1], w1r = wRe[p1], w1i = wIm[p1];
  float Dh = Dp[h];

  float s0r = 0.f, s0i = 0.f, s1r = 0.f, s1i = 0.f;
  const float*    up  = u  + (size_t)b * LL * HH + h;
  unsigned short* yhp = yh + (size_t)b * LL * HH + h;
  unsigned short* ylp = yl + (size_t)b * LL * HH + h;

  for (int l = 0; l < LL; ++l) {
    float uv = up[(size_t)l * HH];
    float t0r = a0r * s0r - a0i * s0i + uv;
    float t0i = a0r * s0i + a0i * s0r;
    float t1r = a1r * s1r - a1i * s1i + uv;
    float t1i = a1r * s1i + a1i * s1r;
    s0r = t0r; s0i = t0i; s1r = t1r; s1i = t1i;

    float p = w0r * s0r - w0i * s0i + w1r * s1r - w1i * s1i;
#pragma unroll
    for (int off = 16; off; off >>= 1) p += __shfl_xor(p, off, 32);

    if (lane == 0) {
      float y = p + Dh * uv;
      float inner = 0.7978845608028654f * (y + 0.044715f * y * y * y);
      float g = 0.5f * y * (1.0f + tanhf(inner));
      unsigned short gh = f32_to_bf16_rn(g);
      yhp[(size_t)l * HH] = gh;
      ylp[(size_t)l * HH] = f32_to_bf16_rn(g - bf16_bits_to_f32(gh));
    }
  }
}

// ---------------------------------------------------------------------------
// Kernel 5: out = skip + bias + act @ W   via v_wmma_f32_16x16x32_bf16,
// 3-term bf16 split (Ah*Bh + Ah*Bl + Al*Bh) for ~f32 accuracy.
// Block tile 128(M) x 64(N), 8 waves, each wave a 16x64 strip (4 C tiles).
// Weight panel staged to LDS by the Tensor Data Mover in 4 K-stages of 128;
// A fragments are direct 16B global vector loads of the pre-split planes.
// ---------------------------------------------------------------------------
__global__ __launch_bounds__(256) void gemm_out_kernel(
    const unsigned short* __restrict__ Ah,    // [M, K] bf16 hi
    const unsigned short* __restrict__ Al,    // [M, K] bf16 lo
    const unsigned short* __restrict__ Bh,    // [N, K] bf16 hi
    const unsigned short* __restrict__ Bl,    // [N, K] bf16 lo
    const float* __restrict__ skip,           // x, [M, N]
    const float* __restrict__ bias,           // [N]
    float* __restrict__ out)                  // [M, N]
{
  __shared__ unsigned short ldsBh[64 * 128];  // 16 KB
  __shared__ unsigned short ldsBl[64 * 128];  // 16 KB

  const int lane  = threadIdx.x & 31;
  const int wave  = threadIdx.x >> 5;
  const int mBase = blockIdx.x * 128 + wave * 16;
  const int nBase = blockIdx.y * 64;

  const int mRow  = lane & 15;   // A layout: M = lane&15
  const int kGrp  = lane >> 4;   // A layout: K group select
  const int nCol  = lane & 15;   // B/C/D layout: N = lane&15
  const int kHalf = lane >> 4;   // B layout: K half select

  v8f zero = {0.f, 0.f, 0.f, 0.f, 0.f, 0.f, 0.f, 0.f};
  v8f acc[4];
#pragma unroll
  for (int j = 0; j < 4; ++j) acc[j] = zero;

  const unsigned short* ahRow = Ah + (size_t)(mBase + mRow) * HH + kGrp * 16;
  const unsigned short* alRow = Al + (size_t)(mBase + mRow) * HH + kGrp * 16;
  const unsigned ldsBhOff = (unsigned)(uintptr_t)(&ldsBh[0]);
  const unsigned ldsBlOff = (unsigned)(uintptr_t)(&ldsBl[0]);

  for (int kb0 = 0; kb0 < HH; kb0 += 128) {
    __syncthreads();   // previous stage's LDS readers done
    if (threadIdx.x < 32) {
      // TDM: 64 rows (nBase..nBase+63) x 128 cols (kb0..kb0+127)
      const unsigned short* gh = Bh + (size_t)nBase * HH + kb0;
      const unsigned short* gl = Bl + (size_t)nBase * HH + kb0;
      tdm_load_2d_bf16(gh, ldsBhOff, HH, HH, 128u, 64u);
      tdm_load_2d_bf16(gl, ldsBlOff, HH, HH, 128u, 64u);
      __builtin_amdgcn_s_wait_tensorcnt(0);
    }
    __syncthreads();   // LDS tile visible to all waves

#pragma unroll
    for (int kk = 0; kk < 128; kk += 32) {
      const int kb = kb0 + kk;
      __builtin_prefetch(ahRow + kb + 32, 0, 1);   // global_prefetch_b8
      // A fragments: two 16B segments per plane (K = kGrp*16+{0..15})
      v8bf a0h = *(const v8bf*)(ahRow + kb);
      v8bf a1h = *(const v8bf*)(ahRow + kb + 8);
      v8bf a0l = *(const v8bf*)(alRow + kb);
      v8bf a1l = *(const v8bf*)(alRow + kb + 8);
      v16bf afh = __builtin_shufflevector(a0h, a1h,
          0, 1, 2, 3, 4, 5, 6, 7, 8, 9, 10, 11, 12, 13, 14, 15);
      v16bf afl = __builtin_shufflevector(a0l, a1l,
          0, 1, 2, 3, 4, 5, 6, 7, 8, 9, 10, 11, 12, 13, 14, 15);

#pragma unroll
      for (int j = 0; j < 4; ++j) {
        const int bo = (j * 16 + nCol) * 128 + kk + kHalf * 16;
        v16bf bh = *(const v16bf*)(ldsBh + bo);   // ds_load_b128 x2
        v16bf bl = *(const v16bf*)(ldsBl + bo);
        acc[j] = __builtin_amdgcn_wmma_f32_16x16x32_bf16(
            false, afh, false, bh, (short)0, acc[j], false, false);
        acc[j] = __builtin_amdgcn_wmma_f32_16x16x32_bf16(
            false, afh, false, bl, (short)0, acc[j], false, false);
        acc[j] = __builtin_amdgcn_wmma_f32_16x16x32_bf16(
            false, afl, false, bh, (short)0, acc[j], false, false);
      }
    }
  }

  // ---- Epilogue: D layout -> rows, add skip + bias, store f32
#pragma unroll
  for (int j = 0; j < 4; ++j) {
    int n = nBase + j * 16 + nCol;
    float bo = bias[n];
#pragma unroll
    for (int r = 0; r < 8; ++r) {
      int m = mBase + (lane >> 4) * 8 + r;   // D layout: M = r + 8*(lane>=16)
      size_t idx = (size_t)m * HH + n;
      out[idx] = skip[idx] + bo + acc[j][r];
    }
  }
}

// ---------------------------------------------------------------------------
// Host-side launcher
// ---------------------------------------------------------------------------
extern "C" void kernel_launch(void* const* d_in, const int* in_sizes, int n_in,
                              void* d_out, int out_size, void* d_ws, size_t ws_size,
                              hipStream_t stream) {
  const float* x       = (const float*)d_in[0];
  const float* Lre     = (const float*)d_in[1];
  const float* Lim     = (const float*)d_in[2];
  const float* log_dt  = (const float*)d_in[3];
  const float* Cre     = (const float*)d_in[4];
  const float* Cim     = (const float*)d_in[5];
  const float* D       = (const float*)d_in[6];
  const float* nscale  = (const float*)d_in[7];
  const float* nbias   = (const float*)d_in[8];
  const float* W_out   = (const float*)d_in[9];
  const float* b_out   = (const float*)d_in[10];
  float* out = (float*)d_out;

  // workspace layout
  char* ws = (char*)d_ws;
  const size_t big  = (size_t)BB * LL * HH * sizeof(float);          // 64 MiB
  const size_t half = (size_t)BB * LL * HH * sizeof(unsigned short); // 32 MiB
  float*          u_norm = (float*)ws;
  unsigned short* yh     = (unsigned short*)(ws + big);
  unsigned short* yl     = (unsigned short*)(ws + big + half);
  float*          aRe    = (float*)(ws + 2 * big);
  float*          aIm    = aRe + HH * NN;
  float*          wRe    = aIm + HH * NN;
  float*          wIm    = wRe + HH * NN;
  unsigned short* Wh     = (unsigned short*)(wIm + HH * NN);
  unsigned short* Wl     = Wh + (size_t)HH * HH;

  // 1) discretized SSM params: H*N / 256 = 128 blocks
  s4d_params_kernel<<<(HH * NN) / 256, 256, 0, stream>>>(
      Lre, Lim, log_dt, Cre, Cim, aRe, aIm, wRe, wIm);

  // 2) weight transpose + bf16 split: H*H / 256 = 1024 blocks
  wsplit_kernel<<<(HH * HH) / 256, 256, 0, stream>>>(W_out, Wh, Wl);

  // 3) layernorm: B*L rows, 8 rows/block -> 4096 blocks
  layernorm_kernel<<<(BB * LL) / 8, 256, 0, stream>>>(x, nscale, nbias, u_norm);

  // 4) SSM scan + gelu (+ bf16 split): B*H waves, 8 waves/block -> 1024 blocks
  s4d_scan_kernel<<<(BB * HH) / 8, 256, 0, stream>>>(
      u_norm, aRe, aIm, wRe, wIm, D, yh, yl);

  // 5) output GEMM + bias + skip: grid (M/128, N/64) = (256, 8)
  gemm_out_kernel<<<dim3(MM / 128, HH / 64), 256, 0, stream>>>(
      yh, yl, Wh, Wl, x, b_out, out);
}
